// RecurTreeGen_48249662603982
// MI455X (gfx1250) — matrix operand order repaired and checked by reading, not compiled, not run on hardware
//
#include <hip/hip_runtime.h>
#include <hip/hip_bf16.h>
#include <math.h>

// Problem constants (match reference)
#define DD    128
#define MM    131072
#define KK    65536      // M/2 partition point
#define ND    640        // 5*D gate width

typedef __attribute__((ext_vector_type(16))) _Float16 v16h;
typedef __attribute__((ext_vector_type(8)))  _Float16 v8h;
typedef __attribute__((ext_vector_type(4)))  _Float16 v4h;
typedef __attribute__((ext_vector_type(8)))  float    v8f;

// ---------------------------------------------------------------------------
// Prep: W [256,640] f32 row-major  ->  Wt [640][256] f16 (transposed).
// A WMMA B-fragment lane load (col n, 16 K-values) becomes one contiguous
// 32-byte read: Wt[n*256 + kbase .. +15].
// ---------------------------------------------------------------------------
__global__ __launch_bounds__(256) void prep_w_kernel(const float* __restrict__ W,
                                                     _Float16* __restrict__ Wt) {
    const int n = blockIdx.x;      // 0..639  (gate column)
    const int k = threadIdx.x;     // 0..255  (input feature)
    Wt[(size_t)n * 256 + k] = (_Float16)W[(size_t)k * ND + n];
}

// ---------------------------------------------------------------------------
// Fused gather + GEMM (WMMA f16->f32) + TreeLSTM epilogue.
// One workgroup = 16 output rows, 8 wave32.
// ---------------------------------------------------------------------------
__global__ __launch_bounds__(256) void tree_lstm_kernel(
    const float* __restrict__ h_bot, const float* __restrict__ c_bot,
    const float* __restrict__ h_buf, const float* __restrict__ c_buf,
    const _Float16* __restrict__ Wt, const float* __restrict__ bias,
    const int* __restrict__ bot_froms0, const int* __restrict__ prev_froms0,
    const int* __restrict__ bot_froms1, const int* __restrict__ prev_froms1,
    float* __restrict__ out_h, float* __restrict__ out_c)
{
    // A tile: 16 rows x 256 K (f16). Row padded 256->264 halfs (528 B = 132
    // dwords, 132 % 64 = 4) so 16 rows at the same K hit distinct banks.
    __shared__ __align__(16) _Float16 As[16][264];
    // Gate tile: 16 rows x 640 f32, padded 640->644 (row delta 4 banks).
    __shared__ __align__(16) float Gs[16][644];

    const int tid   = threadIdx.x;
    const int mBase = blockIdx.x * 16;

    // ---------------- Phase 1: gather h_l | h_r rows, convert to f16 --------
    // 16 rows x 64 float4 (left: q 0..31, right: q 32..63); b128 gathers.
    #pragma unroll
    for (int e = tid; e < 16 * 64; e += 256) {
        const int row   = e >> 6;
        const int q     = e & 63;            // float4 slot within row
        const int mGlob = mBase + row;
        const float* src;
        int d4;
        if (q < 32) {                        // left child h
            d4 = q * 4;
            if (mGlob < KK) src = h_bot + (size_t)bot_froms0[mGlob] * DD;
            else            src = h_buf + (size_t)prev_froms0[mGlob - KK] * DD;
        } else {                             // right child h
            d4 = (q - 32) * 4;
            if (mGlob < KK) src = h_buf + (size_t)prev_froms1[mGlob] * DD;
            else            src = h_bot + (size_t)bot_froms1[mGlob - KK] * DD;
        }
        const float4 v = *(const float4*)(src + d4);
        v4h hv;
        hv.x = (_Float16)v.x; hv.y = (_Float16)v.y;
        hv.z = (_Float16)v.z; hv.w = (_Float16)v.w;
        *(v4h*)&As[row][q * 4] = hv;
    }
    __syncthreads();

    // ---------------- Phase 2: WMMA GEMM, gates -> LDS ----------------------
    const int wave = tid >> 5;
    const int lane = tid & 31;
    const int g    = lane >> 4;   // lane group (0: lanes 0-15, 1: lanes 16-31)
    const int nl   = lane & 15;   // A row / B column within tile

    // Per-wave column tiles: ct = wave + j*8, j = 0..4. Precompute B bases.
    const _Float16* bptr[5];
    #pragma unroll
    for (int j = 0; j < 5; ++j) {
        const int nGlob = (wave + j * 8) * 16 + nl;
        bptr[j] = Wt + (size_t)nGlob * 256 + g * 16;
    }

    v8f acc[5];
    #pragma unroll
    for (int j = 0; j < 5; ++j) acc[j] = (v8f){};

    // K-step outer: one A-fragment LDS load feeds 5 WMMAs.
    #pragma unroll
    for (int ks = 0; ks < 8; ++ks) {
        // A fragment (contiguous K order, matching B): lane holds K
        // {k0..k0+15}, k0 = ks*32 + g*16.
        const int k0 = ks * 32 + g * 16;
        v8h alo = *(const v8h*)&As[nl][k0];
        v8h ahi = *(const v8h*)&As[nl][k0 + 8];
        v16h a  = __builtin_shufflevector(alo, ahi,
                    0,1,2,3,4,5,6,7,8,9,10,11,12,13,14,15);
        #pragma unroll
        for (int j = 0; j < 5; ++j) {
            // B fragment: lane = its column, 16 contiguous halfs (32 B).
            const v16h bf = *(const v16h*)(bptr[j] + ks * 32);
            acc[j] = __builtin_amdgcn_wmma_f32_16x16x32_f16(
                         false, a, false, bf, (short)0, acc[j], false, false);
        }
    }

    // Bias + stage gates in LDS.
    // C/D layout: lane holds one column, rows v + 8*g for VGPR v.
    #pragma unroll
    for (int j = 0; j < 5; ++j) {
        const int nGlob = (wave + j * 8) * 16 + nl;
        const float bv  = bias[nGlob];
        #pragma unroll
        for (int v = 0; v < 8; ++v)
            Gs[v + 8 * g][nGlob] = acc[j][v] + bv;
    }
    __syncthreads();

    // ---------------- Phase 3: gather c rows + TreeLSTM epilogue ------------
    // 16 rows x 32 float4 slots; two iterations of 256 threads.
    #pragma unroll
    for (int e = tid; e < 16 * 32; e += 256) {
        const int row   = e >> 5;
        const int d4    = (e & 31) * 4;
        const int mGlob = mBase + row;

        const float* clsrc;
        const float* crsrc;
        if (mGlob < KK) {
            clsrc = c_bot + (size_t)bot_froms0[mGlob] * DD;
            crsrc = c_buf + (size_t)prev_froms1[mGlob] * DD;
        } else {
            clsrc = c_buf + (size_t)prev_froms0[mGlob - KK] * DD;
            crsrc = c_bot + (size_t)bot_froms1[mGlob - KK] * DD;
        }
        const float4 cl = *(const float4*)(clsrc + d4);
        const float4 cr = *(const float4*)(crsrc + d4);

        float4 hv, cv;
        #pragma unroll
        for (int t = 0; t < 4; ++t) {
            const int d = d4 + t;
            float gi = Gs[row][d];
            float go = Gs[row][DD + d];
            float gu = Gs[row][2 * DD + d];
            float fl = Gs[row][3 * DD + d];
            float fr = Gs[row][4 * DD + d];

            gi = 1.0f / (1.0f + __expf(-gi));
            go = 1.0f / (1.0f + __expf(-go));
            fl = 1.0f / (1.0f + __expf(-fl));
            fr = 1.0f / (1.0f + __expf(-fr));
            gu = tanhf(gu);

            const float clv = (t == 0) ? cl.x : (t == 1) ? cl.y : (t == 2) ? cl.z : cl.w;
            const float crv = (t == 0) ? cr.x : (t == 1) ? cr.y : (t == 2) ? cr.z : cr.w;

            const float c = gi * gu + fl * clv + fr * crv;
            const float h = go * tanhf(c);
            ((float*)&cv)[t] = c;
            ((float*)&hv)[t] = h;
        }
        *(float4*)(out_h + (size_t)mGlob * DD + d4) = hv;
        *(float4*)(out_c + (size_t)mGlob * DD + d4) = cv;
    }
}

extern "C" void kernel_launch(void* const* d_in, const int* in_sizes, int n_in,
                              void* d_out, int out_size, void* d_ws, size_t ws_size,
                              hipStream_t stream) {
    const float* h_bot = (const float*)d_in[0];
    const float* c_bot = (const float*)d_in[1];
    const float* h_buf = (const float*)d_in[2];
    const float* c_buf = (const float*)d_in[3];
    const float* W     = (const float*)d_in[4];
    const float* b     = (const float*)d_in[5];
    const int* bot_froms0  = (const int*)d_in[6];
    // d_in[7]  = bot_tos0  (arange(0,K)   — partition is implicit)
    const int* prev_froms0 = (const int*)d_in[8];
    // d_in[9]  = prev_tos0 (arange(K,M))
    const int* bot_froms1  = (const int*)d_in[10];
    // d_in[11] = bot_tos1  (arange(K,M))
    const int* prev_froms1 = (const int*)d_in[12];
    // d_in[13] = prev_tos1 (arange(0,K))

    _Float16* Wt = (_Float16*)d_ws;              // 640*256*2 = 320 KB scratch

    float* out_h = (float*)d_out;                // h first, then c (ref order)
    float* out_c = out_h + (size_t)MM * DD;

    prep_w_kernel<<<ND, 256, 0, stream>>>(W, Wt);
    tree_lstm_kernel<<<MM / 16, 256, 0, stream>>>(
        h_bot, c_bot, h_buf, c_buf, Wt, b,
        bot_froms0, prev_froms0, bot_froms1, prev_froms1,
        out_h, out_c);
}
